// TransformerAttn_592705487554
// MI455X (gfx1250) — matrix operand order repaired and checked by reading, not compiled, not run on hardware
//
#include <hip/hip_runtime.h>
#include <hip/hip_bf16.h>

typedef _Float16 f16;
typedef __attribute__((ext_vector_type(16))) _Float16 v16h;
typedef __attribute__((ext_vector_type(8)))  _Float16 v8h;
typedef __attribute__((ext_vector_type(8)))  float    v8f;
typedef __attribute__((ext_vector_type(4)))  float    v4f;
typedef __attribute__((ext_vector_type(4)))  unsigned int u32x4;
typedef __attribute__((ext_vector_type(4)))  int      i32x4;
typedef __attribute__((ext_vector_type(8)))  int      i32x8;

#define S_LEN 2048
#define BATCH 32
#define DIM   512

#if defined(__has_builtin)
#  if __has_builtin(__builtin_amdgcn_tensor_load_to_lds)
#    define HAVE_TDM 1
#  endif
#endif
#ifndef HAVE_TDM
#  define HAVE_TDM 0
#endif

__device__ __forceinline__ v8f wmma16(v16h a, v16h b, v8f c) {
    // v_wmma_f32_16x16x32_f16: D = A(16x32) * B(32x16) + C(16x16 f32)
    return __builtin_amdgcn_wmma_f32_16x16x32_f16(false, a, false, b,
                                                  (short)0, c, false, false);
}

__device__ __forceinline__ v8h cvt8(v4f a, v4f b) {
    v8h r;
    r[0] = (f16)a[0]; r[1] = (f16)a[1]; r[2] = (f16)a[2]; r[3] = (f16)a[3];
    r[4] = (f16)b[0]; r[5] = (f16)b[1]; r[6] = (f16)b[2]; r[7] = (f16)b[3];
    return r;
}

// ---------------------------------------------------------------------------
// Kernel 0: one-shot fp32 -> f16 conversion of the three weight matrices so
// the projection GEMM's B-fragments are plain 32B f16 loads (no cvt chain).
// ---------------------------------------------------------------------------
__global__ __launch_bounds__(256)
void wcvt_kernel(const float* __restrict__ W0, const float* __restrict__ W1,
                 const float* __restrict__ W2, f16* __restrict__ dst) {
    const long per = (long)DIM * DIM;
    const long i = ((long)blockIdx.x * 256 + threadIdx.x) * 8;
    const float* src = (i < per)       ? W0 + i
                     : (i < 2 * per)   ? W1 + (i - per)
                                       : W2 + (i - 2 * per);
    *(v8h*)(dst + i) = cvt8(*(const v4f*)src, *(const v4f*)(src + 4));
}

// ---------------------------------------------------------------------------
// Kernel 1: projection GEMM  dst[b*S+s][n] = sum_k seq[s][b][k] * W[n][k] + bias[n]
// Block: 256 threads (8 waves). Tile: 32 rows x 128 cols. Wave w: cols
// [by*128 + w*16, +16), row subtiles m=0,1. transposed!=0 stores dst as
// [b*D + n][s] (keys: makes attention GEMM2 B-frags contiguous).
// ---------------------------------------------------------------------------
__global__ __launch_bounds__(256)
void proj_kernel(const float* __restrict__ seq,
                 const f16* __restrict__ Wh,
                 const float* __restrict__ bias,
                 f16* __restrict__ dst, int transposed) {
    __shared__ f16 Atile[32][DIM + 8];   // f16 seq tile, padded stride

    const int tid  = threadIdx.x;
    const int w    = tid >> 5;
    const int lane = tid & 31;
    const int l16  = lane & 15;
    const int hi   = lane >> 4;          // 0: lanes 0-15, 1: lanes 16-31

    const long g0 = (long)blockIdx.x * 32;      // first global row (b-major)
    const int  b  = (int)(g0 / S_LEN);
    const int  s0 = (int)(g0 % S_LEN);

    // Stage 32x512 fp32 rows -> f16 LDS. 2048 chunks of 8 elements.
    for (int c = tid; c < 32 * 64; c += 256) {
        const int row = c >> 6;
        const int ko  = (c & 63) * 8;
        const float* p = seq + ((long)(s0 + row) * BATCH + b) * DIM + ko;
        *(v8h*)&Atile[row][ko] = cvt8(*(const v4f*)p, *(const v4f*)(p + 4));
    }
    __syncthreads();

    const int n = blockIdx.y * 128 + w * 16 + l16;  // this lane's output column
    v8f acc[2] = {};

    for (int kc = 0; kc < DIM; kc += 32) {
        // B fragment: lane holds column n -> W row n, 16 contiguous k (f16).
        const v16h Bf = *(const v16h*)(Wh + (long)n * DIM + kc + hi * 16);
#pragma unroll
        for (int m = 0; m < 2; ++m) {
            const f16* ap = &Atile[m * 16 + l16][kc + hi * 16];
            union { v16h v; v8h h[2]; } Af;
            Af.h[0] = *(const v8h*)ap;
            Af.h[1] = *(const v8h*)(ap + 8);
            acc[m] = wmma16(Af.v, Bf, acc[m]);
        }
    }

    const float bn = bias[n];
    if (!transposed) {
#pragma unroll
        for (int m = 0; m < 2; ++m)
#pragma unroll
            for (int r = 0; r < 8; ++r) {
                const int s = s0 + m * 16 + hi * 8 + r;
                dst[((long)b * S_LEN + s) * DIM + n] = (f16)(acc[m][r] + bn);
            }
    } else {
#pragma unroll
        for (int m = 0; m < 2; ++m) {
            v8h o;
#pragma unroll
            for (int r = 0; r < 8; ++r) o[r] = (f16)(acc[m][r] + bn);
            f16* p = dst + ((long)b * DIM + n) * S_LEN + s0 + m * 16 + hi * 8;
            *(v8h*)p = o;   // 8 consecutive s positions
        }
    }
}

// ---------------------------------------------------------------------------
// Kernel 2: flash attention. One block (8 waves) per (b, 16-row i-tile).
//   logits[i][j] = value[i] . query[j] * scale  -> online softmax -> @ keys
// Vw/Qw: f16 [b*S+s][d]; Ktw: f16 [b*D+d][s]; out: fp32 [s][b][d].
// Value tile is staged to LDS by the Tensor Data Mover (padded rows).
// ---------------------------------------------------------------------------
__global__ __launch_bounds__(256)
void attn_kernel(const f16* __restrict__ Vw,
                 const f16* __restrict__ Qw,
                 const f16* __restrict__ Ktw,
                 float* __restrict__ out) {
    __shared__ f16   Vt[16][DIM + 8];    // value rows (flash "queries")
    __shared__ float Sb[16][128 + 4];    // logits slab
    __shared__ f16   Pb[16][128 + 8];    // probabilities (A-frag source)
    __shared__ float mrun[16], lrun[16], corr[16];

    const int tid  = threadIdx.x;
    const int w    = tid >> 5;
    const int lane = tid & 31;
    const int l16  = lane & 15;
    const int hi   = lane >> 4;

    const int b  = blockIdx.x >> 7;            // S/16 = 128 i-tiles per batch
    const int i0 = (blockIdx.x & 127) * 16;

    const f16* vbase = Vw + ((long)b * S_LEN + i0) * DIM;

#if HAVE_TDM
    // ---- TDM: DMA the 16x512 f16 value tile into LDS with row padding ----
    if (w == 0) {
        const unsigned lds_base = (unsigned)(unsigned long long)&Vt[0][0];
        const unsigned long long ga = (unsigned long long)vbase;
        u32x4 g0d;
        g0d[0] = 1u;                               // count=1 (valid user D#)
        g0d[1] = lds_base;                         // lds_addr (bytes)
        g0d[2] = (unsigned)ga;                     // global_addr[31:0]
        g0d[3] = (unsigned)((ga >> 32) & 0x1FFFFFFu) | (2u << 30); // [56:32]|type=2
        i32x8 g1d;
        g1d[0] = (1 << 16)                         // data_size = 2 bytes
               | (1 << 20)                         // pad_enable
               | (7 << 22)                         // pad_interval = 256 DWORDs
               | (3 << 25);                        // pad_amount   = 4 DWORDs
        g1d[1] = (int)(512u << 16);                // tensor_dim0 = 512
        g1d[2] = 0;                                // dim0 hi, tensor_dim1 lo (65536)
        g1d[3] = (int)(1u | (512u << 16));         // tensor_dim1 hi=1, tile_dim0=512
        g1d[4] = 16;                               // tile_dim1 = 16, tile_dim2 = 0
        g1d[5] = 512;                              // tensor_dim0_stride = 512
        g1d[6] = 0;
        g1d[7] = 0;
        i32x4 z4 = {0, 0, 0, 0};
#if defined(__clang_major__) && (__clang_major__ >= 23)
        i32x8 z8 = {0, 0, 0, 0, 0, 0, 0, 0};
        __builtin_amdgcn_tensor_load_to_lds(g0d, g1d, z4, z4, z8, 0);
#else
        __builtin_amdgcn_tensor_load_to_lds(g0d, g1d, z4, z4, 0);
#endif
        __builtin_amdgcn_s_wait_tensorcnt(0);
    }
#else
    for (int c = tid; c < 16 * 64; c += 256) {
        const int row = c >> 6;
        const int ko  = (c & 63) * 8;
        *(v8h*)&Vt[row][ko] = *(const v8h*)(vbase + (long)row * DIM + ko);
    }
#endif
    if (tid < 16) { mrun[tid] = -__builtin_huge_valf(); lrun[tid] = 0.0f; }
    __syncthreads();

    v8f o[4] = {};                              // 16 rows x 64 cols (w's strip)
    const float scale = 0.044194173824159216f;  // 1/sqrt(512)

    const f16* qb = Qw + (long)b * S_LEN * DIM;
    const f16* kb = Ktw + (long)b * DIM * S_LEN;

    for (int j0 = 0; j0 < S_LEN; j0 += 128) {
        // ---- GEMM1: this wave's 16x16 logits slab, K = D = 512 ----
        v8f sacc = {};
        const f16* qrow = qb + (long)(j0 + w * 16 + l16) * DIM;
        if (j0 + 128 < S_LEN) {
            __builtin_prefetch(qrow + (long)128 * DIM, 0, 1);      // next j-tile Q
            __builtin_prefetch(kb + (long)(w * 64 + l16) * S_LEN
                               + j0 + 128 + hi * 16, 0, 1);        // next j-tile Kt
        }
        for (int kc = 0; kc < DIM; kc += 32) {
            union { v16h v; v8h h[2]; } Af;
            const f16* ap = &Vt[l16][kc + hi * 16];
            Af.h[0] = *(const v8h*)ap;
            Af.h[1] = *(const v8h*)(ap + 8);
            const v16h Bf = *(const v16h*)(qrow + kc + hi * 16);
            sacc = wmma16(Af.v, Bf, sacc);
        }
#pragma unroll
        for (int r = 0; r < 8; ++r)
            Sb[r + hi * 8][w * 16 + l16] = sacc[r] * scale;
        __syncthreads();

        // ---- online softmax over this 128-wide slab ----
        {
            const int row = tid >> 4;          // 16 threads per row
            const int sub = tid & 15;
            float v[8];
            float mx = -__builtin_huge_valf();
#pragma unroll
            for (int cc = 0; cc < 8; ++cc) {
                v[cc] = Sb[row][sub + cc * 16];
                mx = fmaxf(mx, v[cc]);
            }
#pragma unroll
            for (int m = 1; m < 16; m <<= 1) mx = fmaxf(mx, __shfl_xor(mx, m, 16));
            const float mold = mrun[row];
            const float mnew = fmaxf(mold, mx);
            float sum = 0.0f;
#pragma unroll
            for (int cc = 0; cc < 8; ++cc) {
                const float p = __expf(v[cc] - mnew);
                sum += p;
                Pb[row][sub + cc * 16] = (f16)p;
            }
#pragma unroll
            for (int m = 1; m < 16; m <<= 1) sum += __shfl_xor(sum, m, 16);
            if (sub == 0) {
                const float cr = __expf(mold - mnew);
                corr[row] = cr;
                lrun[row] = lrun[row] * cr + sum;
                mrun[row] = mnew;
            }
        }
        __syncthreads();

        // ---- rescale running output by correction factor ----
        float cr[8];
#pragma unroll
        for (int r = 0; r < 8; ++r) cr[r] = corr[r + hi * 8];
#pragma unroll
        for (int t = 0; t < 4; ++t)
#pragma unroll
            for (int r = 0; r < 8; ++r) o[t][r] *= cr[r];

        // ---- GEMM2: O[:, w*64..] += P(16x128) @ keys(128 x 64) ----
#pragma unroll
        for (int kk = 0; kk < 4; ++kk) {
            union { v16h v; v8h h[2]; } Af;
            const f16* pp = &Pb[l16][kk * 32 + hi * 16];
            Af.h[0] = *(const v8h*)pp;
            Af.h[1] = *(const v8h*)(pp + 8);
#pragma unroll
            for (int t = 0; t < 4; ++t) {
                const int d = w * 64 + t * 16 + l16;
                const f16* kp = kb + (long)d * S_LEN + j0 + kk * 32 + hi * 16;
                const v16h Bf = *(const v16h*)kp;
                o[t] = wmma16(Af.v, Bf, o[t]);
            }
        }
        __syncthreads();
    }

    // ---- epilogue: normalize and store fp32 [s][b][d] ----
    float li[8];
#pragma unroll
    for (int r = 0; r < 8; ++r) li[r] = 1.0f / lrun[r + hi * 8];
#pragma unroll
    for (int t = 0; t < 4; ++t) {
        const int d = w * 64 + t * 16 + l16;
#pragma unroll
        for (int r = 0; r < 8; ++r) {
            const int s = i0 + r + hi * 8;
            out[((long)s * BATCH + b) * DIM + d] = o[t][r] * li[r];
        }
    }
}

// ---------------------------------------------------------------------------
extern "C" void kernel_launch(void* const* d_in, const int* in_sizes, int n_in,
                              void* d_out, int out_size, void* d_ws, size_t ws_size,
                              hipStream_t stream) {
    (void)in_sizes; (void)n_in; (void)out_size; (void)ws_size;
    const float* seq = (const float*)d_in[0];
    const float* Wv  = (const float*)d_in[1];
    const float* bv  = (const float*)d_in[2];
    const float* Wq  = (const float*)d_in[3];
    const float* bq  = (const float*)d_in[4];
    const float* Wk  = (const float*)d_in[5];
    const float* bk  = (const float*)d_in[6];
    float* out = (float*)d_out;

    const size_t mat = (size_t)BATCH * S_LEN * DIM;   // elements per matrix
    const size_t per = (size_t)DIM * DIM;
    f16* Vw  = (f16*)d_ws;           // [b*S+s][d]
    f16* Qw  = Vw + mat;             // [b*S+s][d]
    f16* Ktw = Qw + mat;             // [b*D+d][s]
    f16* Wh  = Ktw + mat;            // f16 copies of Wv|Wq|Wk

    wcvt_kernel<<<dim3(3 * per / (256 * 8)), 256, 0, stream>>>(Wv, Wq, Wk, Wh);

    dim3 gProj(BATCH * S_LEN / 32, DIM / 128);
    proj_kernel<<<gProj, 256, 0, stream>>>(seq, Wh,           bv, Vw, 0);
    proj_kernel<<<gProj, 256, 0, stream>>>(seq, Wh + per,     bq, Qw, 0);
    proj_kernel<<<gProj, 256, 0, stream>>>(seq, Wh + 2 * per, bk, Ktw, 1);

    dim3 gAttn(BATCH * (S_LEN / 16));
    attn_kernel<<<gAttn, 256, 0, stream>>>(Vw, Qw, Ktw, out);
}
